// localUp2_89378269430386
// MI455X (gfx1250) — matrix operand order repaired and checked by reading, not compiled, not run on hardware
//
#include <hip/hip_runtime.h>

// ---------------------------------------------------------------------------
// CDNA5 (gfx1250) implementation of the localUp2 reference:
//   q = W1 @ c1 + b1  (kd=64, C1=512, N=P) ; k = W2 @ c2 + b2 (N=S)
//   energy = q . k[4 bilinear neighbors]; att = softmax*coef
//   res = sum_j out[:, idx_j] * att_j
// GEMM: v_wmma_f32_16x16x32_bf16, double-buffered GLOBAL_LOAD_ASYNC_TO_LDS_B128
// staging (ASYNCcnt-pipelined). W is pre-converted once to bf16 in the exact
// A-fragment swizzle (zero per-step VALU); the streamed B tile is converted
// f32->bf16 exactly once per element via a block-wide LDS restage pass, so
// both A and B fragments are bare 32-byte ds loads feeding the WMMAs.
// ---------------------------------------------------------------------------

typedef __attribute__((ext_vector_type(16))) __bf16 v16bf;
typedef __attribute__((ext_vector_type(2)))  __bf16 v2bf;
typedef __attribute__((ext_vector_type(8)))  float  v8f;
typedef __attribute__((ext_vector_type(4)))  int    v4i_t;

#define KD    64     // kd (M of projection GEMM)
#define C1DIM 512    // K of projection GEMM
#define COUT  256
#define WSWZ_ELEMS (16 * 4 * 32 * 16)   // = KD*C1DIM = 32768 bf16 per W

#if defined(__HIP_DEVICE_COMPILE__) && \
    __has_builtin(__builtin_amdgcn_global_load_async_to_lds_b128)
#define USE_ASYNC_LDS 1
#else
#define USE_ASYNC_LDS 0
#endif

#define ASG __attribute__((address_space(1)))
#define ASL __attribute__((address_space(3)))

__device__ __forceinline__ unsigned short bf16_bits(float f) {
    union { float f; unsigned u; } v; v.f = f;
    unsigned r = v.u + 0x7FFFu + ((v.u >> 16) & 1u);  // round-to-nearest-even
    return (unsigned short)(r >> 16);
}

__device__ __forceinline__ v2bf cvt_pk(float lo, float hi) {
#if defined(__HIP_DEVICE_COMPILE__) && \
    __has_builtin(__builtin_amdgcn_cvt_pk_bf16_f32)
    return __builtin_amdgcn_cvt_pk_bf16_f32(lo, hi);
#else
    // native __bf16 casts: let ISel pick v_cvt_pk_bf16_f32 / v_cvt_bf16_f32
    v2bf r;
    r.x = (__bf16)lo;
    r.y = (__bf16)hi;
    return r;
#endif
}

#if USE_ASYNC_LDS
__device__ __forceinline__ void async_b128(const void* gsrc, const void* lds) {
    __builtin_amdgcn_global_load_async_to_lds_b128(
        (ASG v4i_t*)(unsigned long long)gsrc,
        (ASL v4i_t*)(unsigned)(unsigned long long)lds, 0, 0);
}
__device__ __forceinline__ void wait_async0() {
#if __has_builtin(__builtin_amdgcn_s_wait_asynccnt)
    __builtin_amdgcn_s_wait_asynccnt(0);
#else
    asm volatile("s_wait_asynccnt 0x0" ::: "memory");
#endif
}
#endif

// ---------------------------------------------------------------------------
// prep_wbf16: W [KD][C1] f32 -> bf16 in A-fragment swizzle:
//   dst[((ks*4 + mt)*32 + l)*16 + i] = bf16(W[mt*16 + (l&15)]
//                                            [ks*32 + ((l>>4)<<3) + (i<8?i:8+i)])
// One element per thread (grid 128 x 256 = 32768).
// ---------------------------------------------------------------------------
__global__ __launch_bounds__(256) void prep_wbf16(
    const float* __restrict__ W, unsigned short* __restrict__ dst)
{
    const int o  = blockIdx.x * 256 + threadIdx.x;
    const int i  = o & 15;
    const int l  = (o >> 4) & 31;
    const int mt = (o >> 9) & 3;
    const int ks = o >> 11;
    const int m  = mt * 16 + (l & 15);
    const int k  = ks * 32 + ((l >> 4) << 3) + (i < 8 ? i : 8 + i);
    dst[o] = bf16_bits(W[m * C1DIM + k]);
}

// ---------------------------------------------------------------------------
// proj: Y[n, KD, P] = W[KD, C1] @ X[n, C1, P] + bias[KD]
// block = 256 threads = 8 waves: wave w -> M-tile (w&3), N-pair (w>>2).
// Block tile: 64(M) x 64(N), K loop = 16 steps of 32 (bf16 WMMA).
// ---------------------------------------------------------------------------
__global__ __launch_bounds__(256) void proj_wmma(
    const float* __restrict__ X, const float* __restrict__ W,
    const unsigned short* __restrict__ Wswz,
    const float* __restrict__ bias, float* __restrict__ Y, int P)
{
    const int tid   = threadIdx.x;
    const int lane  = tid & 31;
    const int wave  = tid >> 5;
    const int mt    = wave & 3;          // M tile (16 rows each)
    const int ntp   = wave >> 2;         // N pair: covers cols ntp*32 .. +31
    const int pbase = blockIdx.x * 64;
    const size_t nOff = (size_t)blockIdx.y;

    const float* Xn = X + nOff * C1DIM * (size_t)P;
    float*       Yn = Y + nOff * KD    * (size_t)P;

    v8f acc0 = {};
    v8f acc1 = {};

#if USE_ASYNC_LDS
    // DMA-filled double buffers + one shared bf16 restage tile:
    __shared__ unsigned short ldsAbf[2][4 * 32 * 16];  // A frags  4 KB/buf
    __shared__ float          ldsBf[2][32 * 64];       // B f32    8 KB/buf
    __shared__ unsigned short ldsBbf[32 * 64];         // B bf16   4 KB

    auto issue_tile = [&](int ks, int buf) {
        // A: contiguous 4 KB slice of pre-swizzled W -> 1 b128 per thread
        async_b128(Wswz + ks * (4 * 32 * 16) + tid * 8, &ldsAbf[buf][tid * 8]);
        // B: 32(K) x 64(N) f32 -> 2 b128 per thread; f4 = tid*2+j
        #pragma unroll
        for (int j = 0; j < 2; ++j) {
            const int f4 = tid * 2 + j;
            const int r  = f4 >> 4;
            const int c4 = f4 & 15;
            async_b128(Xn + (size_t)(ks * 32 + r) * P + pbase + c4 * 4,
                       &ldsBf[buf][r * 64 + c4 * 4]);
        }
    };

    issue_tile(0, 0);

    const int aOff = (mt * 32 + lane) * 16;    // A fragment base (ushorts)
    const int bOff = (ntp * 32 + (lane & 15)) * 32 + (lane >> 4) * 16;     // B fragment base (ushorts)
    const int rsR  = tid >> 3;                 // restage row (K)
    const int rsC  = (tid & 7) * 8;            // restage col start

    for (int ks = 0; ks < 16; ++ks) {
        const int cur = ks & 1;
        wait_async0();          // this wave's tile-ks DMA landed in LDS
        __syncthreads();        // tile ks visible; step ks-1 frag reads done
        if (ks + 1 < 16) issue_tile(ks + 1, cur ^ 1);  // overlap with compute

        // restage: convert each B element exactly once (8 floats/thread)
        {
            const float* src = &ldsBf[cur][rsR * 64 + rsC];
            const float4 x0 = *reinterpret_cast<const float4*>(src);
            const float4 x1 = *reinterpret_cast<const float4*>(src + 4);
            union { v2bf h[4]; unsigned short s[8]; } pk;
            pk.h[0] = cvt_pk(x0.x, x0.y); pk.h[1] = cvt_pk(x0.z, x0.w);
            pk.h[2] = cvt_pk(x1.x, x1.y); pk.h[3] = cvt_pk(x1.z, x1.w);
            const int pK = ((rsR >> 3) & 1) * 16 + (rsR & 7) + ((rsR >> 4) << 3);
            #pragma unroll
            for (int j = 0; j < 8; ++j) ldsBbf[(rsC + j) * 32 + pK] = pk.s[j];
        }
        __syncthreads();        // bf16 tile complete

        // fragments: bare 32-byte bf16 reads, zero VALU
        const v16bf fa  = *reinterpret_cast<const v16bf*>(&ldsAbf[cur][aOff]);
        const v16bf fb0 = *reinterpret_cast<const v16bf*>(&ldsBbf[bOff]);
        const v16bf fb1 = *reinterpret_cast<const v16bf*>(&ldsBbf[bOff + 512]);

        acc0 = __builtin_amdgcn_wmma_f32_16x16x32_bf16(
            false, fa, false, fb0, (short)0, acc0, false, false);
        acc1 = __builtin_amdgcn_wmma_f32_16x16x32_bf16(
            false, fa, false, fb1, (short)0, acc1, false, false);
    }
#else
    // ---- fallback: register-staged bf16 tiles --------------------
    (void)Wswz;
    __shared__ unsigned short ldsA[4 * 32 * 16];  // 4 KB (A fragments)
    __shared__ unsigned short ldsB[32 * 64];      // 4 KB (B rows)

    const int aL   = (tid >> 1) & 31;
    const int aMt  = (tid >> 6) & 3;
    const int aRow = aMt * 16 + (aL & 15);
    const int aKofs = ((aL >> 4) << 3) + ((tid & 1) ? 16 : 0);
    const int rB = tid >> 3;
    const int sB = (tid & 7) * 8;

    for (int ks = 0; ks < 16; ++ks) {
        __syncthreads();
        {
            const float* wp = W + aRow * C1DIM + ks * 32 + aKofs;
            float4 w0 = *reinterpret_cast<const float4*>(wp);
            float4 w1 = *reinterpret_cast<const float4*>(wp + 4);
            unsigned short* d = &ldsA[tid * 8];
            d[0] = bf16_bits(w0.x); d[1] = bf16_bits(w0.y);
            d[2] = bf16_bits(w0.z); d[3] = bf16_bits(w0.w);
            d[4] = bf16_bits(w1.x); d[5] = bf16_bits(w1.y);
            d[6] = bf16_bits(w1.z); d[7] = bf16_bits(w1.w);
        }
        {
            const float* sp = Xn + (size_t)(ks * 32 + rB) * P + pbase + sB;
            if (ks + 1 < 16)
                __builtin_prefetch(sp + 32 * (size_t)P, 0, 1);
            float4 x0 = *reinterpret_cast<const float4*>(sp);
            float4 x1 = *reinterpret_cast<const float4*>(sp + 4);
            unsigned short* d = &ldsB[sB * 32 + ((rB >> 3) & 1) * 16 + (rB & 7) + ((rB >> 4) << 3)];
            d[0] = bf16_bits(x0.x); d[32] = bf16_bits(x0.y);
            d[64] = bf16_bits(x0.z); d[96] = bf16_bits(x0.w);
            d[128] = bf16_bits(x1.x); d[160] = bf16_bits(x1.y);
            d[192] = bf16_bits(x1.z); d[224] = bf16_bits(x1.w);
        }
        __syncthreads();

        const v16bf a  = *reinterpret_cast<const v16bf*>(&ldsA[(mt * 32 + lane) * 16]);
        const v16bf b0 = *reinterpret_cast<const v16bf*>(&ldsB[((ntp * 2 + 0) * 16 + (lane & 15)) * 32 + (lane >> 4) * 16]);
        const v16bf b1 = *reinterpret_cast<const v16bf*>(&ldsB[((ntp * 2 + 1) * 16 + (lane & 15)) * 32 + (lane >> 4) * 16]);

        acc0 = __builtin_amdgcn_wmma_f32_16x16x32_bf16(
            false, a, false, b0, (short)0, acc0, false, false);
        acc1 = __builtin_amdgcn_wmma_f32_16x16x32_bf16(
            false, a, false, b1, (short)0, acc1, false, false);
    }
#endif

    // ---- epilogue: C layout -> VGPR r holds M = r + 8*lane[4], N = lane[3:0]
    const int ncol = lane & 15;
    const int mofs = (lane >> 4) << 3;
    #pragma unroll
    for (int r = 0; r < 8; ++r) {
        const int m  = mt * 16 + mofs + r;
        const float bv = bias[m];
        Yn[(size_t)m * P + pbase + (ntp * 2 + 0) * 16 + ncol] = acc0[r] + bv;
        Yn[(size_t)m * P + pbase + (ntp * 2 + 1) * 16 + ncol] = acc1[r] + bv;
    }
}

// ---------------------------------------------------------------------------
// attend: one thread per dest pixel. Replicates the reference bilinear
// index/coef math, 64-dim q.k dots, softmax*coef, 256-channel weighted gather.
// ---------------------------------------------------------------------------
__global__ __launch_bounds__(256) void attend(
    const float* __restrict__ q,    // [n, KD, P]
    const float* __restrict__ kk,   // [n, KD, S]
    const float* __restrict__ outF, // [n, COUT, S]
    float* __restrict__ res,        // [n, COUT, P]
    int hd, int wd, int hs, int ws)
{
    const int P = hd * wd;
    const int S = hs * ws;
    const int p = blockIdx.x * blockDim.x + threadIdx.x;
    if (p >= P) return;
    const size_t nOff = (size_t)blockIdx.y;

    const int row = p / wd;  // dest_X (over hd)
    const int col = p % wd;  // dest_Y (over wd)
    const float scale_h = (float)hs / (float)hd;
    const float scale_w = (float)ws / (float)wd;
    const float src_x = ((float)row + 0.5f) * scale_h - 0.5f;
    const float src_y = ((float)col + 0.5f) * scale_w - 0.5f;

    const int x0 = (int)floorf(src_x);
    const int y0 = (int)floorf(src_y);
    const int x1 = (x0 + 1 < hs - 1) ? x0 + 1 : hs - 1;
    const int y1 = (y0 + 1 < ws - 1) ? y0 + 1 : ws - 1;
    const int x00 = x0 > 0 ? x0 : 0;
    const int y00 = y0 > 0 ? y0 : 0;

    const int i0 = y00 * hs + x00;  // up_left
    const int i1 = y1  * hs + x00;  // up_right
    const int i2 = y00 * hs + x1;   // down_left
    const int i3 = y1  * hs + x1;   // down_right

    float norm = (float)((x1 - x0) * (y1 - y0));
    if (norm == 0.0f) norm = 1.0f;
    const float invn = 1.0f / norm;
    const float dx0 = src_x - (float)x0, dy0 = src_y - (float)y0;
    const float dx1 = (float)x1 - src_x, dy1 = (float)y1 - src_y;
    const float c0 = dx0 * dy0 * invn, c1 = dx0 * dy1 * invn;
    const float c2 = dx1 * dy0 * invn, c3 = dx1 * dy1 * invn;

    // energy = q[:,p] . k[:, idx_j]
    const float* qn = q  + nOff * KD * (size_t)P + p;
    const float* kn = kk + nOff * KD * (size_t)S;
    float e0 = 0.f, e1 = 0.f, e2 = 0.f, e3 = 0.f;
    #pragma unroll 4
    for (int c = 0; c < KD; ++c) {
        const float qv = qn[(size_t)c * P];
        const float* kc = kn + (size_t)c * S;
        e0 += qv * kc[i0];
        e1 += qv * kc[i1];
        e2 += qv * kc[i2];
        e3 += qv * kc[i3];
    }
    const float mx = fmaxf(fmaxf(e0, e1), fmaxf(e2, e3));
    const float w0 = __expf(e0 - mx), w1 = __expf(e1 - mx);
    const float w2 = __expf(e2 - mx), w3 = __expf(e3 - mx);
    const float inv = 1.0f / (w0 + w1 + w2 + w3);
    const float a0 = w0 * inv * c0, a1 = w1 * inv * c1;
    const float a2 = w2 * inv * c2, a3 = w3 * inv * c3;

    const float* on = outF + nOff * COUT * (size_t)S;
    float*       rn = res  + nOff * COUT * (size_t)P + p;
    #pragma unroll 4
    for (int ch = 0; ch < COUT; ++ch) {
        const float* oc = on + (size_t)ch * S;
        rn[(size_t)ch * P] = oc[i0] * a0 + oc[i1] * a1 + oc[i2] * a2 + oc[i3] * a3;
    }
}

// ---------------------------------------------------------------------------
extern "C" void kernel_launch(void* const* d_in, const int* in_sizes, int n_in,
                              void* d_out, int out_size, void* d_ws, size_t ws_size,
                              hipStream_t stream) {
    const float* c1  = (const float*)d_in[0];  // [n, C1, hd, wd]
    const float* c2  = (const float*)d_in[1];  // [n, C1, hs, ws]
    const float* oF  = (const float*)d_in[2];  // [n, Cout, hs, ws]
    const float* W1  = (const float*)d_in[3];  // [kd, C1]
    const float* b1  = (const float*)d_in[4];
    const float* W2  = (const float*)d_in[5];
    const float* b2  = (const float*)d_in[6];
    float*       res = (float*)d_out;          // [n, Cout, hd, wd]

    // Reference setup dims: n=2, C1=512, kd=64, Cout=256, hd=wd=256, hs=ws=128
    const int n  = 2;
    const int hd = 256, wd = 256, hs = 128, ws = 128;
    const int P  = hd * wd;   // 65536
    const int S  = hs * ws;   // 16384

    // workspace: q [n,64,P] | k [n,64,S] | W1 bf16 swizzle | W2 bf16 swizzle
    float* q = (float*)d_ws;
    float* k = q + (size_t)n * KD * P;
    unsigned short* w1bf = (unsigned short*)(k + (size_t)n * KD * S);
    unsigned short* w2bf = w1bf + WSWZ_ELEMS;

    prep_wbf16<<<WSWZ_ELEMS / 256, 256, 0, stream>>>(W1, w1bf);
    prep_wbf16<<<WSWZ_ELEMS / 256, 256, 0, stream>>>(W2, w2bf);

    dim3 gq(P / 64, n);
    proj_wmma<<<gq, 256, 0, stream>>>(c1, W1, w1bf, b1, q, P);
    dim3 gk(S / 64, n);
    proj_wmma<<<gk, 256, 0, stream>>>(c2, W2, w2bf, b2, k, S);
    dim3 ga((P + 255) / 256, n);
    attend<<<ga, 256, 0, stream>>>(q, k, oF, res, hd, wd, hs, ws);
}
